// FlightDelayGNN_56289841381502
// MI455X (gfx1250) — compile-verified, matrix operand order checked by reading, not run on hardware
//
#include <hip/hip_runtime.h>
#include <hip/hip_bf16.h>

// ---------------------------------------------------------------------------
// FlightDelayGNN on MI455X (gfx1250): full pipeline.
// GEMMs: weights pre-packed once per launch into WMMA-fragment-native bf16
// tiles (L2-resident); activations streamed fp32 via b128 loads and converted
// to bf16 in registers; v_wmma_f32_16x16x32_bf16 accumulation in f32.
// Graph scatter/softmax via atomics; BN/ELU/decoder as elementwise kernels.
// ---------------------------------------------------------------------------

typedef __attribute__((ext_vector_type(16))) __bf16 v16bf;
typedef __attribute__((ext_vector_type(8)))  float  v8f;

constexpr int   kN    = 10000;   // nodes
constexpr int   kE    = 320000;  // edges
constexpr int   kFE   = 16;      // edge feats
constexpr int   kHD   = 128;     // hidden
constexpr int   kHC1  = 256;     // HEADS*HD (conv1 out)
constexpr int   kCAT  = 272;     // 2*HD + FE
constexpr float kEPS  = 1e-5f;
constexpr float kSLP  = 0.2f;    // leaky-relu slope

// ---------------- B pre-pack: fragment-native bf16 tiles --------------------
// Packed layout: [kt][nt][lane][j] with 512 bf16 per (kt,nt) tile.
// Fragment layout used here (contiguous k, same order as the A loads):
//   lane 0-15  -> col nt*16+lane,   K = kt*32 + j      (j = 0..15)
//   lane 16-31 -> col nt*16+lane-16, K = kt*32 + 16 + j
// Zero-padded for k >= K so the GEMM tail needs no B-side guard.
__host__ __device__ constexpr size_t packed_elems(int K, int Nc) {
  return (size_t)((K + 31) / 32) * (Nc / 16) * 512;
}

__global__ void pack_b_bf16(const float* __restrict__ B, __bf16* __restrict__ Bp,
                            int K, int Nc) {
  const int ntiles = Nc >> 4;
  const int ktiles = (K + 31) >> 5;
  const size_t total = (size_t)ktiles * ntiles * 512;
  const size_t idx = (size_t)blockIdx.x * blockDim.x + threadIdx.x;
  if (idx >= total) return;
  const int j    = (int)(idx & 15);
  const int lane = (int)((idx >> 4) & 31);
  const size_t t = idx >> 9;
  const int nt = (int)(t % ntiles);
  const int kt = (int)(t / ntiles);
  const int col = nt * 16 + (lane & 15);
  const int k   = kt * 32 + ((lane & 16) ? 16 : 0) + j;
  Bp[idx] = (k < K) ? (__bf16)B[(size_t)k * Nc + col] : (__bf16)0.0f;
}

// ---------------- GEMM: C[M,Nc] = act(A[M,K] @ B[K,Nc] + bias) --------------
// One wave per 16 x (16*NSUB) tile. M multiple of 16 (10000, 320000: yes).
// Main K-loop is branch-free (b128 loads); K%32 tail handled once, padded.
template <int NSUB, bool RELU>
__global__ __launch_bounds__(32)
void wmma_gemm(const float* __restrict__ A, const __bf16* __restrict__ Bp,
               const float* __restrict__ bias, float* __restrict__ C,
               int M, int K, int Nc) {
  const int lane   = threadIdx.x & 31;
  const int m0     = blockIdx.x * 16;
  const int n0     = blockIdx.y * (16 * NSUB);
  const int ntiles = Nc >> 4;
  const int ktiles = (K + 31) >> 5;
  const int kfull  = K >> 5;

  const int row  = m0 + (lane & 15);
  const int koff = (lane & 16) ? 16 : 0;

  const float*  __restrict__ pa = A + (size_t)row * K + koff;          // 16B aligned
  const __bf16* __restrict__ pb = Bp + (((size_t)(n0 >> 4)) * 512) + (size_t)lane * 16;
  const size_t bstride = (size_t)ntiles * 512;

  v8f acc[NSUB] = {};

  for (int kt = 0; kt < kfull; ++kt) {
    const float4* __restrict__ f = reinterpret_cast<const float4*>(pa);
    const float4 f0 = f[0], f1 = f[1], f2 = f[2], f3 = f[3];
    v16bf a;
    a[0]  = (__bf16)f0.x; a[1]  = (__bf16)f0.y; a[2]  = (__bf16)f0.z; a[3]  = (__bf16)f0.w;
    a[4]  = (__bf16)f1.x; a[5]  = (__bf16)f1.y; a[6]  = (__bf16)f1.z; a[7]  = (__bf16)f1.w;
    a[8]  = (__bf16)f2.x; a[9]  = (__bf16)f2.y; a[10] = (__bf16)f2.z; a[11] = (__bf16)f2.w;
    a[12] = (__bf16)f3.x; a[13] = (__bf16)f3.y; a[14] = (__bf16)f3.z; a[15] = (__bf16)f3.w;
#pragma unroll
    for (int t = 0; t < NSUB; ++t) {
      const v16bf b = *reinterpret_cast<const v16bf*>(pb + (size_t)t * 512);
      acc[t] = __builtin_amdgcn_wmma_f32_16x16x32_bf16(
          false, a, false, b, (short)0, acc[t], false, false);
    }
    pa += 32;
    pb += bstride;
  }

  if (kfull < ktiles) {   // padded tail (K%32 != 0); runs at most once
    const int kb = kfull * 32 + koff;
    v16bf a;
#pragma unroll
    for (int j = 0; j < 16; ++j) {
      const int k = kb + j;
      a[j] = (k < K) ? (__bf16)A[(size_t)row * K + k] : (__bf16)0.0f;
    }
#pragma unroll
    for (int t = 0; t < NSUB; ++t) {
      const v16bf b = *reinterpret_cast<const v16bf*>(pb + (size_t)t * 512);
      acc[t] = __builtin_amdgcn_wmma_f32_16x16x32_bf16(
          false, a, false, b, (short)0, acc[t], false, false);
    }
  }

  // C/D layout: lane l -> col (l&15); VGPR r -> row r (+8 for lanes>=16).
  const int colb  = lane & 15;
  const int rbase = (lane & 16) ? 8 : 0;
#pragma unroll
  for (int t = 0; t < NSUB; ++t) {
    const int col = n0 + 16 * t + colb;
    const float bv = bias ? bias[col] : 0.0f;
#pragma unroll
    for (int r = 0; r < 8; ++r) {
      float v = acc[t][r] + bv;
      if (RELU) v = v > 0.0f ? v : 0.0f;
      C[(size_t)(m0 + rbase + r) * Nc + col] = v;
    }
  }
}

// ---------------- order-preserving float<->uint for atomicMax ---------------
__device__ inline unsigned f32_ord(float f) {
  unsigned u = __float_as_uint(f);
  return (u & 0x80000000u) ? ~u : (u | 0x80000000u);
}
__device__ inline float ord_f32(unsigned o) {
  unsigned u = (o & 0x80000000u) ? (o & 0x7fffffffu) : ~o;
  return __uint_as_float(u);
}

// ---------------- GATv2 edge pipeline ---------------------------------------
__global__ void edge_logits_kernel(const int* __restrict__ src, const int* __restrict__ dst,
                                   const float* __restrict__ xl, const float* __restrict__ xr,
                                   const float* __restrict__ ee, const float* __restrict__ att,
                                   float* __restrict__ logit, int E, int H, int C) {
  const int t = blockIdx.x * blockDim.x + threadIdx.x;
  if (t >= E * H) return;
  const int e = t / H, h = t - e * H;
  const float* __restrict__ pl = xl + ((size_t)src[e] * H + h) * C;
  const float* __restrict__ pr = xr + ((size_t)dst[e] * H + h) * C;
  const float* __restrict__ pe = ee + ((size_t)e * H + h) * C;
  const float* __restrict__ pa = att + (size_t)h * C;
  float acc = 0.0f;
  for (int c = 0; c < C; ++c) {
    float v = pl[c] + pr[c] + pe[c];
    v = v > 0.0f ? v : kSLP * v;
    acc += v * pa[c];
  }
  logit[t] = acc;
}

__global__ void seg_max_kernel(const float* __restrict__ logit, const int* __restrict__ dst,
                               unsigned* __restrict__ mx, int E, int H) {
  const int t = blockIdx.x * blockDim.x + threadIdx.x;
  if (t >= E * H) return;
  const int e = t / H, h = t - e * H;
  atomicMax(&mx[(size_t)dst[e] * H + h], f32_ord(logit[t]));
}

__global__ void seg_expsum_kernel(float* __restrict__ logit, const int* __restrict__ dst,
                                  const unsigned* __restrict__ mx, float* __restrict__ denom,
                                  int E, int H) {
  const int t = blockIdx.x * blockDim.x + threadIdx.x;
  if (t >= E * H) return;
  const int e = t / H, h = t - e * H;
  const float m = ord_f32(mx[(size_t)dst[e] * H + h]);
  const float a = expf(logit[t] - m);
  logit[t] = a;
  atomicAdd(&denom[(size_t)dst[e] * H + h], a);
}

__global__ void seg_aggregate_kernel(const float* __restrict__ alpha, const float* __restrict__ denom,
                                     const int* __restrict__ src, const int* __restrict__ dst,
                                     const float* __restrict__ xl, float* __restrict__ agg,
                                     int E, int H, int C) {
  const int t = blockIdx.x;           // e*H + h
  const int e = t / H, h = t - e * H;
  const int c = threadIdx.x;
  const float an = alpha[t] / denom[(size_t)dst[e] * H + h];
  atomicAdd(&agg[((size_t)dst[e] * H + h) * C + c],
            an * xl[((size_t)src[e] * H + h) * C + c]);
}

// ---------------- BatchNorm (biased var) + ELU ------------------------------
__global__ void col_stats_kernel(const float* __restrict__ X, const float* __restrict__ bias,
                                 int M, int Ncols, float* __restrict__ mu, float* __restrict__ rsig) {
  __shared__ float sh_s[256];
  __shared__ float sh_q[256];
  const int c = blockIdx.x;
  const float bv = bias ? bias[c] : 0.0f;
  float s = 0.0f, q = 0.0f;
  for (int r = threadIdx.x; r < M; r += blockDim.x) {
    const float v = X[(size_t)r * Ncols + c] + bv;
    s += v; q += v * v;
  }
  sh_s[threadIdx.x] = s; sh_q[threadIdx.x] = q;
  __syncthreads();
  for (int w = 128; w > 0; w >>= 1) {
    if ((int)threadIdx.x < w) {
      sh_s[threadIdx.x] += sh_s[threadIdx.x + w];
      sh_q[threadIdx.x] += sh_q[threadIdx.x + w];
    }
    __syncthreads();
  }
  if (threadIdx.x == 0) {
    const float m   = sh_s[0] / (float)M;
    const float var = sh_q[0] / (float)M - m * m;
    mu[c]   = m;
    rsig[c] = rsqrtf(var + kEPS);
  }
}

__global__ void bn_elu_kernel(const float* __restrict__ X, const float* __restrict__ bias,
                              const float* __restrict__ mu, const float* __restrict__ rsig,
                              const float* __restrict__ g, const float* __restrict__ beta,
                              float* __restrict__ Y, int M, int Ncols) {
  const size_t idx = (size_t)blockIdx.x * blockDim.x + threadIdx.x;
  if (idx >= (size_t)M * Ncols) return;
  const int c = (int)(idx % Ncols);
  const float bv = bias ? bias[c] : 0.0f;
  float v = (X[idx] + bv - mu[c]) * rsig[c] * g[c] + beta[c];
  Y[idx] = v > 0.0f ? v : expm1f(v);   // ELU(alpha=1)
}

// ---------------- decoder helpers -------------------------------------------
__global__ void build_cat_kernel(const float* __restrict__ h, const int* __restrict__ src,
                                 const int* __restrict__ dst, const float* __restrict__ ea,
                                 float* __restrict__ cat, int E) {
  const size_t idx = (size_t)blockIdx.x * blockDim.x + threadIdx.x;
  if (idx >= (size_t)E * kCAT) return;
  const int e = (int)(idx / kCAT);
  const int c = (int)(idx % kCAT);
  float v;
  if (c < kHD)            v = h[(size_t)src[e] * kHD + c];
  else if (c < 2 * kHD)   v = h[(size_t)dst[e] * kHD + (c - kHD)];
  else                    v = ea[(size_t)e * kFE + (c - 2 * kHD)];
  cat[idx] = v;
}

__global__ void final_dot_kernel(const float* __restrict__ z2, const float* __restrict__ Wd3,
                                 const float* __restrict__ bd3, float* __restrict__ out, int E) {
  const int e = blockIdx.x * blockDim.x + threadIdx.x;
  if (e >= E) return;
  float acc = bd3[0];
#pragma unroll
  for (int c = 0; c < 32; ++c) acc += z2[(size_t)e * 32 + c] * Wd3[c];
  out[e] = acc;
}

// ---------------- host-side launch helpers ----------------------------------
static inline void pack_weight(const float* B, __bf16* Bp, int K, int Nc, hipStream_t s) {
  const size_t total = packed_elems(K, Nc);
  pack_b_bf16<<<(int)((total + 255) / 256), 256, 0, s>>>(B, Bp, K, Nc);
}

static inline void launch_gemm(const float* A, const __bf16* Bp, const float* bias, float* C,
                               int M, int K, int Nc, bool relu, hipStream_t s) {
  const dim3 wave(32);
  if (Nc % 64 == 0) {
    const dim3 g(M / 16, Nc / 64);
    if (relu) wmma_gemm<4, true ><<<g, wave, 0, s>>>(A, Bp, bias, C, M, K, Nc);
    else      wmma_gemm<4, false><<<g, wave, 0, s>>>(A, Bp, bias, C, M, K, Nc);
  } else {  // Nc == 32
    const dim3 g(M / 16, Nc / 32);
    if (relu) wmma_gemm<2, true ><<<g, wave, 0, s>>>(A, Bp, bias, C, M, K, Nc);
    else      wmma_gemm<2, false><<<g, wave, 0, s>>>(A, Bp, bias, C, M, K, Nc);
  }
}

// ---------------------------------------------------------------------------
extern "C" void kernel_launch(void* const* d_in, const int* in_sizes, int n_in,
                              void* d_out, int out_size, void* d_ws, size_t ws_size,
                              hipStream_t stream) {
  (void)in_sizes; (void)n_in; (void)out_size; (void)ws_size;

  const float* x     = (const float*)d_in[0];
  const int*   ei    = (const int*)  d_in[1];
  const float* ea    = (const float*)d_in[2];
  const float* Wn    = (const float*)d_in[3];
  const float* bn_b  = (const float*)d_in[4];
  const float* Wep   = (const float*)d_in[5];
  const float* bep   = (const float*)d_in[6];
  const float* Wl1   = (const float*)d_in[7];
  const float* bl1   = (const float*)d_in[8];
  const float* Wr1   = (const float*)d_in[9];
  const float* br1   = (const float*)d_in[10];
  const float* Wee1  = (const float*)d_in[11];
  const float* att1  = (const float*)d_in[12];
  const float* bias1 = (const float*)d_in[13];
  const float* Wl2   = (const float*)d_in[14];
  const float* bl2   = (const float*)d_in[15];
  const float* Wr2   = (const float*)d_in[16];
  const float* br2   = (const float*)d_in[17];
  const float* Wee2  = (const float*)d_in[18];
  const float* att2  = (const float*)d_in[19];
  const float* bias2 = (const float*)d_in[20];
  const float* g1    = (const float*)d_in[21];
  const float* beta1 = (const float*)d_in[22];
  const float* g2    = (const float*)d_in[23];
  const float* beta2 = (const float*)d_in[24];
  const float* Wd1   = (const float*)d_in[25];
  const float* bd1   = (const float*)d_in[26];
  const float* Wd2   = (const float*)d_in[27];
  const float* bd2   = (const float*)d_in[28];
  const float* Wd3   = (const float*)d_in[29];
  const float* bd3   = (const float*)d_in[30];

  const int* src = ei;          // edge_index[0]
  const int* dst = ei + kE;     // edge_index[1]
  float* out = (float*)d_out;

  // Workspace bump allocator (256B aligned). Peak use ~755 MB with region reuse.
  char* base = (char*)d_ws;
  size_t off = 0;
  auto alloc = [&](size_t bytes) -> void* {
    void* p = (void*)(base + off);
    off = (off + bytes + 255) & ~(size_t)255;
    return p;
  };
  auto allocf = [&](size_t nf) -> float* { return (float*)alloc(nf * sizeof(float)); };
  auto allocp = [&](int K, int Nc) -> __bf16* {
    return (__bf16*)alloc(packed_elems(K, Nc) * sizeof(__bf16));
  };

  // packed weights (~3 MB total, L2-resident)
  __bf16* pWn   = allocp(kN,   kHD);
  __bf16* pWep  = allocp(kFE,  kHD);
  __bf16* pWl1  = allocp(kHD,  kHC1);
  __bf16* pWr1  = allocp(kHD,  kHC1);
  __bf16* pWee1 = allocp(kHD,  kHC1);
  __bf16* pWl2  = allocp(kHC1, kHD);
  __bf16* pWr2  = allocp(kHC1, kHD);
  __bf16* pWee2 = allocp(kHD,  kHD);
  __bf16* pWd1  = allocp(kCAT, 64);
  __bf16* pWd2  = allocp(64,   32);

  float* h0     = allocf((size_t)kN * kHD);
  float* e_emb  = allocf((size_t)kE * kHD);
  float* xl1    = allocf((size_t)kN * kHC1);
  float* xr1    = allocf((size_t)kN * kHC1);
  float* regA   = allocf((size_t)kE * kCAT);     // ee1 (E*256) then cat (E*272)
  float* ee1    = regA;
  float* cat    = regA;
  float* logit1 = allocf((size_t)kE * 2);        // becomes alpha1 in place
  unsigned* mx1 = (unsigned*)allocf((size_t)kN * 2);
  float* denom1 = allocf((size_t)kN * 2);
  float* agg1   = allocf((size_t)kN * kHC1);
  float* mu1    = allocf(kHC1);
  float* rs1    = allocf(kHC1);
  float* h1     = allocf((size_t)kN * kHC1);
  float* xl2    = allocf((size_t)kN * kHD);
  float* xr2    = allocf((size_t)kN * kHD);
  float* regB   = allocf((size_t)kE * kHD);      // ee2 (E*128) then z1 (E*64) + z2 (E*32)
  float* ee2    = regB;
  float* z1     = regB;
  float* z2     = regB + (size_t)kE * 64;
  float* logit2 = allocf((size_t)kE);
  unsigned* mx2 = (unsigned*)allocf((size_t)kN);
  float* denom2 = allocf((size_t)kN);
  float* agg2   = allocf((size_t)kN * kHD);
  float* mu2    = allocf(kHD);
  float* rs2    = allocf(kHD);
  float* h2     = allocf((size_t)kN * kHD);

  // ---- pack all weights (cheap; runs every call: deterministic) --------
  pack_weight(Wn,   pWn,   kN,   kHD,  stream);
  pack_weight(Wep,  pWep,  kFE,  kHD,  stream);
  pack_weight(Wl1,  pWl1,  kHD,  kHC1, stream);
  pack_weight(Wr1,  pWr1,  kHD,  kHC1, stream);
  pack_weight(Wee1, pWee1, kHD,  kHC1, stream);
  pack_weight(Wl2,  pWl2,  kHC1, kHD,  stream);
  pack_weight(Wr2,  pWr2,  kHC1, kHD,  stream);
  pack_weight(Wee2, pWee2, kHD,  kHD,  stream);
  pack_weight(Wd1,  pWd1,  kCAT, 64,   stream);
  pack_weight(Wd2,  pWd2,  64,   32,   stream);

  // ---- encoders --------------------------------------------------------
  launch_gemm(x,  pWn,  bn_b, h0,    kN, kN,  kHD, false, stream);  // big one
  launch_gemm(ea, pWep, bep,  e_emb, kE, kFE, kHD, false, stream);

  // ---- conv1: GATv2 (heads=2, concat) ----------------------------------
  launch_gemm(h0,    pWl1,  bl1,     xl1, kN, kHD, kHC1, false, stream);
  launch_gemm(h0,    pWr1,  br1,     xr1, kN, kHD, kHC1, false, stream);
  launch_gemm(e_emb, pWee1, nullptr, ee1, kE, kHD, kHC1, false, stream);

  hipMemsetAsync(mx1, 0, (size_t)kN * 2 * sizeof(unsigned), stream);  // 0 < ord(-inf)
  hipMemsetAsync(denom1, 0, (size_t)kN * 2 * sizeof(float), stream);
  hipMemsetAsync(agg1, 0, (size_t)kN * kHC1 * sizeof(float), stream);

  edge_logits_kernel<<<(kE * 2 + 255) / 256, 256, 0, stream>>>(
      src, dst, xl1, xr1, ee1, att1, logit1, kE, 2, kHD);
  seg_max_kernel<<<(kE * 2 + 255) / 256, 256, 0, stream>>>(logit1, dst, mx1, kE, 2);
  seg_expsum_kernel<<<(kE * 2 + 255) / 256, 256, 0, stream>>>(logit1, dst, mx1, denom1, kE, 2);
  seg_aggregate_kernel<<<kE * 2, kHD, 0, stream>>>(
      logit1, denom1, src, dst, xl1, agg1, kE, 2, kHD);

  col_stats_kernel<<<kHC1, 256, 0, stream>>>(agg1, bias1, kN, kHC1, mu1, rs1);
  bn_elu_kernel<<<(int)(((size_t)kN * kHC1 + 255) / 256), 256, 0, stream>>>(
      agg1, bias1, mu1, rs1, g1, beta1, h1, kN, kHC1);

  // ---- conv2: GATv2 (heads=1, mean == identity) ------------------------
  launch_gemm(h1,    pWl2,  bl2,     xl2, kN, kHC1, kHD, false, stream);
  launch_gemm(h1,    pWr2,  br2,     xr2, kN, kHC1, kHD, false, stream);
  launch_gemm(e_emb, pWee2, nullptr, ee2, kE, kHD,  kHD, false, stream);

  hipMemsetAsync(mx2, 0, (size_t)kN * sizeof(unsigned), stream);
  hipMemsetAsync(denom2, 0, (size_t)kN * sizeof(float), stream);
  hipMemsetAsync(agg2, 0, (size_t)kN * kHD * sizeof(float), stream);

  edge_logits_kernel<<<(kE + 255) / 256, 256, 0, stream>>>(
      src, dst, xl2, xr2, ee2, att2, logit2, kE, 1, kHD);
  seg_max_kernel<<<(kE + 255) / 256, 256, 0, stream>>>(logit2, dst, mx2, kE, 1);
  seg_expsum_kernel<<<(kE + 255) / 256, 256, 0, stream>>>(logit2, dst, mx2, denom2, kE, 1);
  seg_aggregate_kernel<<<kE, kHD, 0, stream>>>(
      logit2, denom2, src, dst, xl2, agg2, kE, 1, kHD);

  col_stats_kernel<<<kHD, 256, 0, stream>>>(agg2, bias2, kN, kHD, mu2, rs2);
  bn_elu_kernel<<<(int)(((size_t)kN * kHD + 255) / 256), 256, 0, stream>>>(
      agg2, bias2, mu2, rs2, g2, beta2, h2, kN, kHD);

  // ---- decoder MLP ------------------------------------------------------
  build_cat_kernel<<<(int)(((size_t)kE * kCAT + 255) / 256), 256, 0, stream>>>(
      h2, src, dst, ea, cat, kE);
  launch_gemm(cat, pWd1, bd1, z1, kE, kCAT, 64, true, stream);
  launch_gemm(z1,  pWd2, bd2, z2, kE, 64,   32, true, stream);
  final_dot_kernel<<<(kE + 255) / 256, 256, 0, stream>>>(z2, Wd3, bd3, out, kE);
}
